// Mamba3LikeScan_25357486916250
// MI455X (gfx1250) — matrix operand order repaired and checked by reading, not compiled, not run on hardware
//
#include <hip/hip_runtime.h>
#include <hip/hip_bf16.h>
#include <math.h>

// Problem constants (B,S,D,N) = (2,2048,1024,16)
#define BB 2
#define SS 2048
#define DD 1024
#define NN 16
#define TT (BB*SS)       // 4096 tokens
#define DN (DD*NN)       // 16384

typedef __attribute__((ext_vector_type(16))) __bf16 bf16x16;
typedef __attribute__((ext_vector_type(8)))  __bf16 bf16x8;
typedef __attribute__((ext_vector_type(8)))  float  f32x8;

enum { EPI_BC = 0, EPI_SOFTPLUS = 1, EPI_SIGMOID = 2, EPI_RESID = 3 };

// Non-temporal bf16 helpers (keep the 268 MB b/c streams out of L2 so the
// 73 MB of bf16 weights stay resident in the 192 MB global L2 between GEMMs).
static __device__ __forceinline__ void nt_store_bf16(__bf16* p, float v) {
    __bf16 h = (__bf16)v;
    unsigned short u;
    __builtin_memcpy(&u, &h, 2);
    __builtin_nontemporal_store(u, (unsigned short*)p);
}
static __device__ __forceinline__ float nt_load_bf16(const __bf16* p) {
    unsigned short u = __builtin_nontemporal_load((const unsigned short*)p);
    __bf16 h;
    __builtin_memcpy(&h, &u, 2);
    return (float)h;
}

// ---------------------------------------------------------------------------
// fp32 -> bf16 conversion (weights)
// ---------------------------------------------------------------------------
__global__ __launch_bounds__(256)
void cvt_f32_bf16_kernel(const float* __restrict__ src, __bf16* __restrict__ dst, size_t n) {
    size_t i = (size_t)blockIdx.x * blockDim.x + threadIdx.x;
    if (i < n) dst[i] = (__bf16)src[i];
}

// ---------------------------------------------------------------------------
// LayerNorm over D=1024; one block per token; writes fp32 + bf16 copies
// ---------------------------------------------------------------------------
__global__ __launch_bounds__(256)
void ln_kernel(const float* __restrict__ x, const float* __restrict__ w,
               const float* __restrict__ b, float* __restrict__ nf32,
               __bf16* __restrict__ nbf16) {
    const int row = blockIdx.x;
    const float* xr = x + (size_t)row * DD;
    float v[4];
    float s = 0.f, s2 = 0.f;
#pragma unroll
    for (int i = 0; i < 4; ++i) {
        v[i] = xr[threadIdx.x + i * 256];
        s += v[i];
        s2 += v[i] * v[i];
    }
#pragma unroll
    for (int off = 16; off; off >>= 1) {
        s  += __shfl_xor(s,  off, 32);
        s2 += __shfl_xor(s2, off, 32);
    }
    __shared__ float ps[8], ps2[8];
    const int wv = threadIdx.x >> 5;
    if ((threadIdx.x & 31) == 0) { ps[wv] = s; ps2[wv] = s2; }
    __syncthreads();
    if (wv == 0) {
        float a  = (threadIdx.x < 8) ? ps[threadIdx.x]  : 0.f;
        float a2 = (threadIdx.x < 8) ? ps2[threadIdx.x] : 0.f;
#pragma unroll
        for (int off = 4; off; off >>= 1) {
            a  += __shfl_xor(a,  off, 32);
            a2 += __shfl_xor(a2, off, 32);
        }
        if (threadIdx.x == 0) { ps[0] = a; ps2[0] = a2; }
    }
    __syncthreads();
    const float mu   = ps[0] * (1.f / DD);
    const float var  = ps2[0] * (1.f / DD) - mu * mu;
    const float rstd = rsqrtf(var + 1e-5f);
#pragma unroll
    for (int i = 0; i < 4; ++i) {
        const int c = threadIdx.x + i * 256;
        const float nv = (v[i] - mu) * rstd * w[c] + b[c];
        nf32 [(size_t)row * DD + c] = nv;
        nbf16[(size_t)row * DD + c] = (__bf16)nv;
    }
}

// ---------------------------------------------------------------------------
// bf16 WMMA GEMM:  Out[m,n] = sum_k A[m,k] * W[n,k]  (+bias, epilogue mode)
// A: [M,K] bf16 row-major, W: [Nout,K] bf16 row-major.
// Wave tile: 32(M) x 64(N) = 2x4 grid of 16x16 WMMAs (B fragment reused
// across both M halves). Block = 8 waves as 4(M) x 2(N) -> 128 x 128 tile.
// Fragment layouts: K contiguous per lane half (k = 16*(lane/16) + s) in A and B (wave32, 16-bit A 16x32, B 32x16).
// ---------------------------------------------------------------------------
__global__ __launch_bounds__(256)
void gemm_bf16_kernel(const __bf16* __restrict__ A, const __bf16* __restrict__ W,
                      const float* __restrict__ bias, const float* __restrict__ resid,
                      void* __restrict__ out, int M, int Nout, int K, int mode) {
    const int lane = threadIdx.x;       // 0..31
    const int wave = threadIdx.y;       // 0..7
    const int wm = wave & 3;            // M sub-tile
    const int wn = wave >> 2;           // N sub-tile
    const int m0 = blockIdx.y * 128 + wm * 32;
    const int n0 = blockIdx.x * 128 + wn * 64;
    const int l16 = lane & 15;
    const int kh  = lane >> 4;          // K-half selector

    f32x8 acc[2][4] = {};
    const __bf16* arow0 = A + (size_t)(m0 + l16) * K;
    const __bf16* arow1 = A + (size_t)(m0 + 16 + l16) * K;

    for (int k0 = 0; k0 < K; k0 += 32) {
        // A fragments (16x32 each): lane row = l16; elems 0-15 = K[k0+kh*16..],
        // contiguous (matches B fragment)
        union { bf16x16 v; bf16x8 h[2]; } af[2];
        af[0].h[0] = *(const bf16x8*)(arow0 + k0 + kh * 16);
        af[0].h[1] = *(const bf16x8*)(arow0 + k0 + kh * 16 + 8);
        af[1].h[0] = *(const bf16x8*)(arow1 + k0 + kh * 16);
        af[1].h[1] = *(const bf16x8*)(arow1 + k0 + kh * 16 + 8);
#pragma unroll
        for (int j = 0; j < 4; ++j) {
            // B fragment (32x16): lane col = l16, 16 contiguous K at k0+kh*16
            bf16x16 bfrag =
                *(const bf16x16*)(W + (size_t)(n0 + j * 16 + l16) * K + k0 + kh * 16);
#pragma unroll
            for (int i = 0; i < 2; ++i) {
                acc[i][j] = __builtin_amdgcn_wmma_f32_16x16x32_bf16(
                    false, af[i].v, false, bfrag, (short)0, acc[i][j], false, false);
            }
        }
    }

    // Epilogue. C/D layout: VGPR r, lanes 0-15 -> M = r, lanes 16-31 -> M = 8+r.
#pragma unroll
    for (int j = 0; j < 4; ++j) {
        const int col = n0 + j * 16 + l16;
        const float bv = bias[col];
#pragma unroll
        for (int i = 0; i < 2; ++i) {
#pragma unroll
            for (int r = 0; r < 8; ++r) {
                const int row = m0 + i * 16 + r + kh * 8;
                const size_t idx = (size_t)row * Nout + col;
                float v = acc[i][j][r] + bv;
                if (mode == EPI_BC) {
                    nt_store_bf16(((__bf16*)out) + idx, v);   // streamed, bypass L2 RT
                } else if (mode == EPI_SOFTPLUS) {
                    float sp = (v > 20.f) ? v : log1pf(expf(v));
                    ((float*)out)[idx] = sp + 0.001f;
                } else if (mode == EPI_SIGMOID) {
                    ((float*)out)[idx] = 1.f / (1.f + expf(-v));
                } else { // EPI_RESID
                    ((float*)out)[idx] = resid[idx] + v;
                }
            }
        }
    }
}

// ---------------------------------------------------------------------------
// Sequential scan over S. One thread per (b,d,n); n = lane%16 so the N=16
// reduction is 4 shfl_xor steps within a wave32 half-group. Writes gate*y bf16.
// b/c terms are streamed with non-temporal loads (read-once data).
// ---------------------------------------------------------------------------
__global__ __launch_bounds__(256)
void scan_kernel(const __bf16* __restrict__ bterm, const __bf16* __restrict__ cterm,
                 const float* __restrict__ delta, const float* __restrict__ gate,
                 const float* __restrict__ nf32, const unsigned char* __restrict__ mask,
                 const float* __restrict__ log_a, const float* __restrict__ skip,
                 __bf16* __restrict__ gys) {
    const int tid = blockIdx.x * 256 + threadIdx.x;
    const int n = tid & (NN - 1);
    const int d = (tid >> 4) & (DD - 1);
    const int b = tid >> 14;

    const float la = log_a[d * NN + n];
    const float sp = (la > 20.f) ? la : log1pf(expf(la));
    const float a  = -sp - 1e-4f;               // negative
    const float neg_inv = 1.f / (-a + 1e-4f);
    const float sk = skip[d];

    float state = 0.f;
    for (int t = 0; t < SS; ++t) {
        const int row = b * SS + t;
        const size_t bc = (size_t)row * DN + d * NN + n;
        const float dt = delta[(size_t)row * DD + d];
        const float alpha = expf(dt * a);
        const float beta  = (1.f - alpha) * neg_inv;
        const float bt = nt_load_bf16(bterm + bc);
        const float upd = alpha * state + beta * bt;
        state = mask[row] ? upd : state;
        float prod = nt_load_bf16(cterm + bc) * state;
#pragma unroll
        for (int off = 8; off; off >>= 1) prod += __shfl_xor(prod, off, 32);
        if (n == 0) {
            const size_t rd = (size_t)row * DD + d;
            const float y = prod + sk * nf32[rd];
            gys[rd] = (__bf16)(gate[rd] * y);
        }
    }
}

// ---------------------------------------------------------------------------
// Host launcher
// ---------------------------------------------------------------------------
extern "C" void kernel_launch(void* const* d_in, const int* in_sizes, int n_in,
                              void* d_out, int out_size, void* d_ws, size_t ws_size,
                              hipStream_t stream) {
    const float* x       = (const float*)d_in[0];
    const unsigned char* mask = (const unsigned char*)d_in[1];
    const float* norm_w  = (const float*)d_in[2];
    const float* norm_b  = (const float*)d_in[3];
    const float* delta_w = (const float*)d_in[4];
    const float* delta_b = (const float*)d_in[5];
    const float* b_w     = (const float*)d_in[6];
    const float* b_b     = (const float*)d_in[7];
    const float* c_w     = (const float*)d_in[8];
    const float* c_b     = (const float*)d_in[9];
    const float* gate_w  = (const float*)d_in[10];
    const float* gate_b  = (const float*)d_in[11];
    const float* out_w   = (const float*)d_in[12];
    const float* out_b   = (const float*)d_in[13];
    const float* skip    = (const float*)d_in[14];
    const float* log_a   = (const float*)d_in[15];

    // Workspace carve-up
    char* ws = (char*)d_ws;
    size_t off = 0;
    auto take = [&](size_t bytes) -> char* {
        char* p = ws + off;
        off = (off + bytes + 255) & ~(size_t)255;
        return p;
    };
    __bf16* nbf16   = (__bf16*)take((size_t)TT * DD * 2);
    __bf16* gys     = (__bf16*)take((size_t)TT * DD * 2);
    float*  nf32    = (float*) take((size_t)TT * DD * 4);
    float*  delta   = (float*) take((size_t)TT * DD * 4);
    float*  gate    = (float*) take((size_t)TT * DD * 4);
    __bf16* wdelta  = (__bf16*)take((size_t)DD * DD * 2);
    __bf16* wgate   = (__bf16*)take((size_t)DD * DD * 2);
    __bf16* wout    = (__bf16*)take((size_t)DD * DD * 2);
    __bf16* wb      = (__bf16*)take((size_t)DN * DD * 2);
    __bf16* wc      = (__bf16*)take((size_t)DN * DD * 2);
    __bf16* bterm   = (__bf16*)take((size_t)TT * DN * 2);
    __bf16* cterm   = (__bf16*)take((size_t)TT * DN * 2);
    (void)ws_size; (void)n_in; (void)in_sizes; (void)out_size;

    // 1) Weight conversions fp32 -> bf16
    auto cvt = [&](const float* s, __bf16* dst, size_t n) {
        cvt_f32_bf16_kernel<<<(unsigned)((n + 255) / 256), 256, 0, stream>>>(s, dst, n);
    };
    cvt(delta_w, wdelta, (size_t)DD * DD);
    cvt(gate_w,  wgate,  (size_t)DD * DD);
    cvt(out_w,   wout,   (size_t)DD * DD);
    cvt(b_w,     wb,     (size_t)DN * DD);
    cvt(c_w,     wc,     (size_t)DN * DD);

    // 2) LayerNorm
    ln_kernel<<<TT, 256, 0, stream>>>(x, norm_w, norm_b, nf32, nbf16);

    // 3) Projections (WMMA bf16, fp32 accumulate). Block tile 128x128.
    dim3 blk(32, 8);
    gemm_bf16_kernel<<<dim3(DD / 128, TT / 128), blk, 0, stream>>>(
        nbf16, wdelta, delta_b, nullptr, delta, TT, DD, DD, EPI_SOFTPLUS);
    gemm_bf16_kernel<<<dim3(DD / 128, TT / 128), blk, 0, stream>>>(
        nbf16, wgate, gate_b, nullptr, gate, TT, DD, DD, EPI_SIGMOID);
    gemm_bf16_kernel<<<dim3(DN / 128, TT / 128), blk, 0, stream>>>(
        nbf16, wb, b_b, nullptr, bterm, TT, DN, DD, EPI_BC);
    gemm_bf16_kernel<<<dim3(DN / 128, TT / 128), blk, 0, stream>>>(
        nbf16, wc, c_b, nullptr, cterm, TT, DN, DD, EPI_BC);

    // 4) Sequential scan over S; produces gate*y in bf16
    scan_kernel<<<(BB * DD * NN) / 256, 256, 0, stream>>>(
        bterm, cterm, delta, gate, nf32, mask, log_a, skip, gys);

    // 5) Output projection + residual: d_out = x + (gate*ys) @ out_w.T + out_b
    gemm_bf16_kernel<<<dim3(DD / 128, TT / 128), blk, 0, stream>>>(
        gys, wout, out_b, x, (float*)d_out, TT, DD, DD, EPI_RESID);
}
